// MaskedSelfAttention_5617817223361
// MI455X (gfx1250) — matrix operand order repaired and checked by reading, not compiled, not run on hardware
//
#include <hip/hip_runtime.h>
#include <hip/hip_fp16.h>

typedef __attribute__((ext_vector_type(16))) _Float16 v16h;
typedef __attribute__((ext_vector_type(8)))  _Float16 v8h;
typedef __attribute__((ext_vector_type(8)))  float    v8f;

#define WMMA_F16(a, b, c) \
  __builtin_amdgcn_wmma_f32_16x16x32_f16(false, (a), false, (b), (short)0, (c), false, false)

#define T_SEQ   165
#define TPAD    192
#define C_EMB   384
#define NHEAD   6
#define HDIM    64
#define BATCH   256
#define M_TOT   (BATCH * T_SEQ)        // 42240
#define LDS_STR 200                    // padded f16 stride: 400B (16B aligned, bank-conflict free)

static __device__ __forceinline__ v8f zero_v8f() {
  v8f z;
#pragma unroll
  for (int i = 0; i < 8; ++i) z[i] = 0.0f;
  return z;
}

// A-fragment (16x32, f16): lane L holds row L&15.
// ISA layout: lanes 0-15 -> K = kbase+{0..7, 16..23}; lanes 16-31 -> K = kbase+{8..15, 24..31}.
static __device__ __forceinline__ v16h load_a_frag(const _Float16* base, int ld, int lane, int kbase) {
  const int r = lane & 15, sub = lane >> 4;
  const _Float16* p = base + (size_t)r * ld + kbase + sub * 8;
  v8h lo = *(const v8h*)(p);
  v8h hi = *(const v8h*)(p + 16);
  v16h a;
#pragma unroll
  for (int i = 0; i < 8; ++i) { a[i] = lo[i]; a[8 + i] = hi[i]; }
  return a;
}

// B-fragment (32x16, f16) from a TRANSPOSED operand (row n holds K values of output column n).
// ISA layout: lanes 0-15 -> K = kbase+{0..7, 16..23}; lanes 16-31 -> K = kbase+{8..15, 24..31}.
static __device__ __forceinline__ v16h load_b_frag(const _Float16* baseT, int ld, int lane, int kbase) {
  const int n = lane & 15, sub = lane >> 4;
  const _Float16* p = baseT + (size_t)n * ld + kbase + sub * 8;
  v8h lo = *(const v8h*)(p);
  v8h hi = *(const v8h*)(p + 16);
  v16h b;
#pragma unroll
  for (int i = 0; i < 8; ++i) { b[i] = lo[i]; b[8 + i] = hi[i]; }
  return b;
}

// ---------------- prep kernels ----------------
__global__ __launch_bounds__(256) void cvt_f32_f16_kernel(const float* __restrict__ in,
                                                          _Float16* __restrict__ out, int n) {
  int i = blockIdx.x * 256 + threadIdx.x;
  if (i < n) out[i] = (_Float16)in[i];
}

// w[K][N] (f32) -> wt[N][K] (f16)
__global__ __launch_bounds__(256) void transpose_cvt_kernel(const float* __restrict__ w,
                                                            _Float16* __restrict__ wt,
                                                            int K, int N) {
  int i = blockIdx.x * 256 + threadIdx.x;
  if (i < K * N) {
    int kx = i / N, n = i - kx * N;
    wt[(size_t)n * K + kx] = (_Float16)w[i];
  }
}

// ---------------- QKV GEMM: [M_TOT x 384] @ Wt[1152 x 384]^T + bias -> q/k/v f16 [B,H,TPAD,64] ----
__global__ __launch_bounds__(256) void qkv_gemm_kernel(const _Float16* __restrict__ A,
                                                       const _Float16* __restrict__ Wt,
                                                       const float* __restrict__ bias,
                                                       _Float16* __restrict__ qb,
                                                       _Float16* __restrict__ kb,
                                                       _Float16* __restrict__ vb) {
  const int lane = threadIdx.x & 31, wave = threadIdx.x >> 5;
  const int wm = wave & 3, wn = wave >> 2;
  const int mbase = blockIdx.x * 128 + wm * 32;
  const int nbase = blockIdx.y * 64 + wn * 32;

  v8f acc[2][2];
  acc[0][0] = zero_v8f(); acc[0][1] = zero_v8f();
  acc[1][0] = zero_v8f(); acc[1][1] = zero_v8f();

  for (int kk = 0; kk < C_EMB; kk += 32) {
    v16h a0 = load_a_frag(A + (size_t)mbase * C_EMB, C_EMB, lane, kk);
    v16h a1 = load_a_frag(A + (size_t)(mbase + 16) * C_EMB, C_EMB, lane, kk);
    v16h b0 = load_b_frag(Wt + (size_t)nbase * C_EMB, C_EMB, lane, kk);
    v16h b1 = load_b_frag(Wt + (size_t)(nbase + 16) * C_EMB, C_EMB, lane, kk);
    acc[0][0] = WMMA_F16(a0, b0, acc[0][0]);
    acc[0][1] = WMMA_F16(a0, b1, acc[0][1]);
    acc[1][0] = WMMA_F16(a1, b0, acc[1][0]);
    acc[1][1] = WMMA_F16(a1, b1, acc[1][1]);
  }

  const int nlo = lane & 15, hf = lane >> 4;
#pragma unroll
  for (int tn = 0; tn < 2; ++tn) {
    const int c = nbase + tn * 16 + nlo;           // 0..1151
    const float bv = bias[c];
    const int which = c / C_EMB;                   // 0=q 1=k 2=v
    const int col = c - which * C_EMB;
    const int h = col >> 6, d = col & 63;
    _Float16* dst = (which == 0) ? qb : (which == 1) ? kb : vb;
#pragma unroll
    for (int tm = 0; tm < 2; ++tm) {
#pragma unroll
      for (int j = 0; j < 8; ++j) {
        const int row = mbase + tm * 16 + j + 8 * hf;   // < 42240 always
        const int bidx = row / T_SEQ;
        const int t = row - bidx * T_SEQ;
        dst[(((size_t)(bidx * NHEAD + h)) * TPAD + t) * HDIM + d] =
            (_Float16)(acc[tm][tn][j] + bv);
      }
    }
  }
}

// ---------------- fused attention: one block per (b,h) ----------------
__global__ __launch_bounds__(256) void attn_kernel(const _Float16* __restrict__ qb,
                                                   const _Float16* __restrict__ kb,
                                                   const _Float16* __restrict__ vb,
                                                   _Float16* __restrict__ yh) {
  __shared__ _Float16 s_vT[HDIM * LDS_STR];          // v transposed: row d over t
  __shared__ _Float16 s_attn[8 * 16 * LDS_STR];      // per-wave P tile (16 x TPAD)

  const int bh = blockIdx.x;
  const int b = bh / NHEAD, h = bh - b * NHEAD;
  const _Float16* qh = qb + (size_t)bh * TPAD * HDIM;
  const _Float16* kh = kb + (size_t)bh * TPAD * HDIM;
  const _Float16* vh = vb + (size_t)bh * TPAD * HDIM;

  // build v^T in LDS; zero the padded rows so poisoned ws can't inject NaN*0
  for (int i = threadIdx.x; i < TPAD * HDIM; i += 256) {
    const int t = i >> 6, d = i & 63;
    s_vT[d * LDS_STR + t] = (t < T_SEQ) ? vh[i] : (_Float16)0.0f;
  }
  __syncthreads();

  const int lane = threadIdx.x & 31, wave = threadIdx.x >> 5;
  const int nlo = lane & 15, hf = lane >> 4;
  _Float16* aw = &s_attn[wave * 16 * LDS_STR];

  for (int qt = wave; qt < 11; qt += 8) {            // 11 q-tiles cover 165(+pad) rows
    // ---- S = (q @ k^T) for all 12 key tiles ----
    v16h aq0 = load_a_frag(qh + (size_t)qt * 16 * HDIM, HDIM, lane, 0);
    v16h aq1 = load_a_frag(qh + (size_t)qt * 16 * HDIM, HDIM, lane, 32);
    v8f sc[12];
#pragma unroll
    for (int kt = 0; kt < 12; ++kt) {
      v16h bk0 = load_b_frag(kh + (size_t)kt * 16 * HDIM, HDIM, lane, 0);
      v16h bk1 = load_b_frag(kh + (size_t)kt * 16 * HDIM, HDIM, lane, 32);
      v8f c = zero_v8f();
      c = WMMA_F16(aq0, bk0, c);
      c = WMMA_F16(aq1, bk1, c);
      sc[kt] = c;
    }
    // ---- masked softmax (row = vgpr j + 8*hf, col = kt*16 + nlo) ----
#pragma unroll
    for (int j = 0; j < 8; ++j) {
      const int t = qt * 16 + j + 8 * hf;
      float mx = -__builtin_inff();
#pragma unroll
      for (int kt = 0; kt < 12; ++kt) {
        const int n = kt * 16 + nlo;
        const bool allowed = (n < T_SEQ) && ((t != T_SEQ - 1) || (n == T_SEQ - 1));
        const float vs = allowed ? sc[kt][j] * 0.125f : -__builtin_inff();
        sc[kt][j] = vs;
        mx = fmaxf(mx, vs);
      }
#pragma unroll
      for (int m = 1; m <= 8; m <<= 1) mx = fmaxf(mx, __shfl_xor(mx, m, 32));
      float sum = 0.0f;
#pragma unroll
      for (int kt = 0; kt < 12; ++kt) {
        const float p = __expf(sc[kt][j] - mx);
        sc[kt][j] = p;
        sum += p;
      }
#pragma unroll
      for (int m = 1; m <= 8; m <<= 1) sum += __shfl_xor(sum, m, 32);
      const float inv = 1.0f / sum;
#pragma unroll
      for (int kt = 0; kt < 12; ++kt) sc[kt][j] *= inv;
    }
    // ---- P (C layout) -> LDS row-major so it can be re-read as an A fragment ----
#pragma unroll
    for (int kt = 0; kt < 12; ++kt) {
#pragma unroll
      for (int j = 0; j < 8; ++j)
        aw[(j + 8 * hf) * LDS_STR + kt * 16 + nlo] = (_Float16)sc[kt][j];
    }
    // ---- Y = P @ V  (K = TPAD = 6 chunks of 32) ----
    v8f yacc[4];
#pragma unroll
    for (int dt = 0; dt < 4; ++dt) yacc[dt] = zero_v8f();
#pragma unroll
    for (int kc = 0; kc < 6; ++kc) {
      v16h ap = load_a_frag(aw, LDS_STR, lane, kc * 32);
#pragma unroll
      for (int dt = 0; dt < 4; ++dt) {
        v16h bv = load_b_frag(&s_vT[dt * 16 * LDS_STR], LDS_STR, lane, kc * 32);
        yacc[dt] = WMMA_F16(ap, bv, yacc[dt]);
      }
    }
    // ---- store y as f16 into [M_TOT, 384] for the projection GEMM ----
#pragma unroll
    for (int dt = 0; dt < 4; ++dt) {
#pragma unroll
      for (int j = 0; j < 8; ++j) {
        const int t = qt * 16 + j + 8 * hf;
        if (t < T_SEQ) {
          const size_t row = (size_t)b * T_SEQ + t;
          const int c = h * HDIM + dt * 16 + nlo;
          yh[row * C_EMB + c] = (_Float16)yacc[dt][j];
        }
      }
    }
  }
}

// ---------------- projection GEMM: [M_TOT x 384] @ Wt[384 x 384]^T + bias -> f32 out ----------
__global__ __launch_bounds__(256) void proj_gemm_kernel(const _Float16* __restrict__ A,
                                                        const _Float16* __restrict__ Wt,
                                                        const float* __restrict__ bias,
                                                        float* __restrict__ out) {
  const int lane = threadIdx.x & 31, wave = threadIdx.x >> 5;
  const int wm = wave & 3, wn = wave >> 2;
  const int mbase = blockIdx.x * 128 + wm * 32;
  const int nbase = blockIdx.y * 64 + wn * 32;

  v8f acc[2][2];
  acc[0][0] = zero_v8f(); acc[0][1] = zero_v8f();
  acc[1][0] = zero_v8f(); acc[1][1] = zero_v8f();

  for (int kk = 0; kk < C_EMB; kk += 32) {
    v16h a0 = load_a_frag(A + (size_t)mbase * C_EMB, C_EMB, lane, kk);
    v16h a1 = load_a_frag(A + (size_t)(mbase + 16) * C_EMB, C_EMB, lane, kk);
    v16h b0 = load_b_frag(Wt + (size_t)nbase * C_EMB, C_EMB, lane, kk);
    v16h b1 = load_b_frag(Wt + (size_t)(nbase + 16) * C_EMB, C_EMB, lane, kk);
    acc[0][0] = WMMA_F16(a0, b0, acc[0][0]);
    acc[0][1] = WMMA_F16(a0, b1, acc[0][1]);
    acc[1][0] = WMMA_F16(a1, b0, acc[1][0]);
    acc[1][1] = WMMA_F16(a1, b1, acc[1][1]);
  }

  const int nlo = lane & 15, hf = lane >> 4;
#pragma unroll
  for (int tn = 0; tn < 2; ++tn) {
    const int c = nbase + tn * 16 + nlo;
    const float bv = bias[c];
#pragma unroll
    for (int tm = 0; tm < 2; ++tm) {
#pragma unroll
      for (int j = 0; j < 8; ++j) {
        const int row = mbase + tm * 16 + j + 8 * hf;
        out[(size_t)row * C_EMB + c] = acc[tm][tn][j] + bv;
      }
    }
  }
}

// ---------------- launcher ----------------
extern "C" void kernel_launch(void* const* d_in, const int* in_sizes, int n_in,
                              void* d_out, int out_size, void* d_ws, size_t ws_size,
                              hipStream_t stream) {
  const float* x      = (const float*)d_in[0];
  const float* w_qkv  = (const float*)d_in[1];
  const float* b_qkv  = (const float*)d_in[2];
  const float* w_proj = (const float*)d_in[3];
  const float* b_proj = (const float*)d_in[4];
  float* out = (float*)d_out;

  char* base = (char*)d_ws;
  size_t off = 0;
  auto take = [&](size_t elems) -> _Float16* {
    _Float16* p = (_Float16*)(base + off);
    off += (elems * sizeof(_Float16) + 255) & ~(size_t)255;
    return p;
  };
  _Float16* x_h    = take((size_t)M_TOT * C_EMB);                 // 32.4 MB
  _Float16* wt_qkv = take((size_t)(3 * C_EMB) * C_EMB);           // 0.88 MB
  _Float16* wt_prj = take((size_t)C_EMB * C_EMB);                 // 0.29 MB
  _Float16* qb     = take((size_t)BATCH * NHEAD * TPAD * HDIM);   // 37.7 MB
  _Float16* kb     = take((size_t)BATCH * NHEAD * TPAD * HDIM);
  _Float16* vb     = take((size_t)BATCH * NHEAD * TPAD * HDIM);
  _Float16* y_h    = take((size_t)M_TOT * C_EMB);                 // 32.4 MB

  // 1) f32 -> f16 conversions / weight transposes
  {
    const int n = M_TOT * C_EMB;
    cvt_f32_f16_kernel<<<(n + 255) / 256, 256, 0, stream>>>(x, x_h, n);
  }
  transpose_cvt_kernel<<<(C_EMB * 3 * C_EMB + 255) / 256, 256, 0, stream>>>(w_qkv, wt_qkv, C_EMB, 3 * C_EMB);
  transpose_cvt_kernel<<<(C_EMB * C_EMB + 255) / 256, 256, 0, stream>>>(w_proj, wt_prj, C_EMB, C_EMB);

  // 2) QKV GEMM (WMMA): block tile 128x64, wave tile 32x32
  qkv_gemm_kernel<<<dim3(M_TOT / 128, (3 * C_EMB) / 64), 256, 0, stream>>>(x_h, wt_qkv, b_qkv, qb, kb, vb);

  // 3) fused attention: one block per (batch, head)
  attn_kernel<<<dim3(BATCH * NHEAD), 256, 0, stream>>>(qb, kb, vb, y_h);

  // 4) projection GEMM (WMMA) with f32 output + bias
  proj_gemm_kernel<<<dim3(M_TOT / 128, C_EMB / 64), 256, 0, stream>>>(y_h, wt_prj, b_proj, out);
}
